// DualGatedRecurrence_51728586113466
// MI455X (gfx1250) — matrix operand order repaired and checked by reading, not compiled, not run on hardware
//
#include <hip/hip_runtime.h>
#include <hip/hip_bf16.h>

// ---------------------------------------------------------------------------
// DualGatedRecurrence for MI455X (gfx1250, wave32, WMMA)
// B=4, T=2048, D=TOT=1024 -> BT=8192 rows, merged width M2=2048.
// ---------------------------------------------------------------------------

#define BT    8192
#define DIM   1024
#define M2    2048
#define NCHK  32          // scan chunks
#define TCHK  64          // T / NCHK
#define BD    4096        // B * DIM channels per scan

typedef __attribute__((ext_vector_type(16))) __bf16 v16bf;
typedef __attribute__((ext_vector_type(8)))  float  v8f;

union FragU { v16bf v; unsigned int w[8]; };

__device__ __forceinline__ unsigned short f2bf(float f) {
    unsigned int u = __builtin_bit_cast(unsigned int, f);
    unsigned int r = u + 0x7FFFu + ((u >> 16) & 1u);   // round-to-nearest-even
    return (unsigned short)(r >> 16);
}

__device__ __forceinline__ float sigf(float x) {
    return 1.0f / (1.0f + __expf(-x));
}

// ---------------------------------------------------------------------------
// hproj on one adjacent pair: the reference fwht butterflies indices (2i,2i+1)
// at EVERY stage, so the whole 4x(10-stage) transform is pair-local.
// ---------------------------------------------------------------------------
__device__ __forceinline__ void hproj_pair(float a, float b,
                                           const float* __restrict__ sc, int d,
                                           float& o0, float& o1) {
    float h0 = a * sc[d];
    float h1 = b * sc[d + 1];
#pragma unroll
    for (int i = 1; i < 5; ++i) {
#pragma unroll
        for (int st = 0; st < 10; ++st) {
            float t0 = h0 + h1;
            float t1 = h0 - h1;
            h0 = t0; h1 = t1;
        }
        h0 *= 0.03125f * sc[i * DIM + d];       // 0.03125 = 1024^-0.5
        h1 *= 0.03125f * sc[i * DIM + d + 1];
    }
    o0 = h0; o1 = h1;
}

// ---------------------------------------------------------------------------
// Weight transpose + fp32->bf16 convert:  dst[c*R + r] = bf16(src[r*C + c])
// ---------------------------------------------------------------------------
__global__ __launch_bounds__(256) void k_transpose_bf16(
        const float* __restrict__ src, unsigned short* __restrict__ dst,
        int R, int C) {
    __shared__ float tile[32][33];
    int lx = threadIdx.x & 31;
    int ly = threadIdx.x >> 5;
    int c0 = blockIdx.x * 32, r0 = blockIdx.y * 32;
#pragma unroll
    for (int i = 0; i < 4; ++i) {
        int r = ly + i * 8;
        tile[r][lx] = src[(size_t)(r0 + r) * C + c0 + lx];
    }
    __syncthreads();
#pragma unroll
    for (int i = 0; i < 4; ++i) {
        int cc = ly + i * 8;
        dst[(size_t)(c0 + cc) * R + r0 + lx] = f2bf(tile[lx][cc]);
    }
}

// ---------------------------------------------------------------------------
// Pointwise: 4 hprojs + gates, bf16 copy of x. One thread per adjacent pair.
// ---------------------------------------------------------------------------
__global__ __launch_bounds__(256) void k_pointwise(
        const float* __restrict__ x,
        const float* __restrict__ fgs, const float* __restrict__ fvs,
        const float* __restrict__ sgs, const float* __restrict__ svs,
        unsigned short* __restrict__ xbf,
        float* __restrict__ ff, float* __restrict__ vf,
        float* __restrict__ fs, float* __restrict__ vs) {
    int gid = blockIdx.x * 256 + threadIdx.x;       // 0 .. BT*DIM/2-1
    int p   = gid & 511;
    int row = gid >> 9;
    int d   = p * 2;
    size_t base = (size_t)row * DIM + d;

    float2 xv = *(const float2*)(x + base);
    float g0, g1, w0, w1;

    hproj_pair(xv.x, xv.y, fgs, d, g0, g1);
    hproj_pair(xv.x, xv.y, fvs, d, w0, w1);
    float f0 = sigf(g0 + 0.5f);                     // GATE_LB_FAST = 0
    float f1 = sigf(g1 + 0.5f);
    *(float2*)(ff + base) = make_float2(f0, f1);
    *(float2*)(vf + base) = make_float2(w0, w1);

    hproj_pair(xv.x, xv.y, sgs, d, g0, g1);
    hproj_pair(xv.x, xv.y, svs, d, w0, w1);
    f0 = 0.85f + 0.15f * sigf(g0 + 0.5f);           // GATE_LB_SLOW = 0.85
    f1 = 0.85f + 0.15f * sigf(g1 + 0.5f);
    *(float2*)(fs + base) = make_float2(f0, f1);
    *(float2*)(vs + base) = make_float2(w0, w1);

    unsigned int pk = (unsigned int)f2bf(xv.x) | ((unsigned int)f2bf(xv.y) << 16);
    *(unsigned int*)(xbf + base) = pk;
}

// ---------------------------------------------------------------------------
// Chunked scan pass 1 (float2 over channel pairs): A = prod f, H = local scan.
// thread -> (scan, chunk c, pair pr) ; pair = 2 adjacent d channels.
// ---------------------------------------------------------------------------
__global__ __launch_bounds__(256) void k_chunk(
        const float* __restrict__ ff, const float* __restrict__ vf,
        const float* __restrict__ fs, const float* __restrict__ vs,
        float* __restrict__ chA, float* __restrict__ chH) {
    int tid  = blockIdx.x * 256 + threadIdx.x;      // 0 .. 2*2048*NCHK-1
    int pr   = tid & 2047;
    int c    = (tid >> 11) & (NCHK - 1);
    int scan = tid >> 16;
    const float* F = scan ? fs : ff;
    const float* V = scan ? vs : vf;
    int b  = pr >> 9;
    int dp = (pr & 511) * 2;
    size_t base = (size_t)b * 2048 * DIM + (size_t)(c * TCHK) * DIM + dp;
    float2 A = make_float2(1.0f, 1.0f);
    float2 H = make_float2(0.0f, 0.0f);
    for (int t = 0; t < TCHK; ++t) {
        float2 fv = *(const float2*)(F + base);
        float2 vv = *(const float2*)(V + base);
        A.x *= fv.x;                      A.y *= fv.y;
        H.x = fv.x * H.x + (1.0f - fv.x) * vv.x;
        H.y = fv.y * H.y + (1.0f - fv.y) * vv.y;
        base += DIM;
    }
    int o = scan * (NCHK * BD) + c * BD + b * DIM + dp;
    *(float2*)(chA + o) = A;
    *(float2*)(chH + o) = H;
}

// Pass 2: serial prefix over chunks -> carry entering each chunk.
__global__ __launch_bounds__(256) void k_prefix(
        const float* __restrict__ chA, const float* __restrict__ chH,
        float* __restrict__ carry) {
    int tid  = blockIdx.x * 256 + threadIdx.x;      // 0 .. 2*2048-1
    int pr   = tid & 2047;
    int scan = tid >> 11;
    int ch   = (pr >> 9) * DIM + (pr & 511) * 2;
    float2 h = make_float2(0.0f, 0.0f);
#pragma unroll
    for (int c = 0; c < NCHK; ++c) {
        int o = scan * (NCHK * BD) + c * BD + ch;
        *(float2*)(carry + o) = h;
        float2 A = *(const float2*)(chA + o);
        float2 H = *(const float2*)(chH + o);
        h.x = A.x * h.x + H.x;
        h.y = A.y * h.y + H.y;
    }
}

// Pass 3: rescan each chunk from its carry, writing h_t.
__global__ __launch_bounds__(256) void k_scanout(
        const float* __restrict__ ff, const float* __restrict__ vf,
        const float* __restrict__ fs, const float* __restrict__ vs,
        const float* __restrict__ carry,
        float* __restrict__ hf, float* __restrict__ hs) {
    int tid  = blockIdx.x * 256 + threadIdx.x;
    int pr   = tid & 2047;
    int c    = (tid >> 11) & (NCHK - 1);
    int scan = tid >> 16;
    const float* F = scan ? fs : ff;
    const float* V = scan ? vs : vf;
    float*       O = scan ? hs : hf;
    int b  = pr >> 9;
    int dp = (pr & 511) * 2;
    size_t base = (size_t)b * 2048 * DIM + (size_t)(c * TCHK) * DIM + dp;
    float2 h = *(const float2*)(carry + scan * (NCHK * BD) + c * BD + b * DIM + dp);
    for (int t = 0; t < TCHK; ++t) {
        float2 fv = *(const float2*)(F + base);
        float2 vv = *(const float2*)(V + base);
        h.x = fv.x * h.x + (1.0f - fv.x) * vv.x;
        h.y = fv.y * h.y + (1.0f - fv.y) * vv.y;
        *(float2*)(O + base) = h;
        base += DIM;
    }
}

__global__ __launch_bounds__(256) void k_zero(float* __restrict__ p, int n) {
    int i = blockIdx.x * 256 + threadIdx.x;
    if (i < n) p[i] = 0.0f;
}

__global__ __launch_bounds__(256) void k_rs(float* __restrict__ rowss) {
    int i = blockIdx.x * 256 + threadIdx.x;
    rowss[i] = rsqrtf(rowss[i] * (1.0f / (float)M2) + 1e-6f);
}

// ---------------------------------------------------------------------------
// bf16 WMMA GEMM core: block tile 128x128, 8 waves (2x4), K-step 32.
// Register double-buffered staging; B tile stored K-pair-interleaved so each
// B-fragment word is a single ds_load_b32 (no VALU repacking at consume time).
//   As: 128 rows x 40 ushort  (32 cols + pad, 16B-aligned rows)
//   Bp: 16 k-pairs x 132 uint (128 cols + pad; 132 % 64 = 4 -> no conflicts)
// ---------------------------------------------------------------------------
__device__ __forceinline__ void gemm_core(
        const unsigned short* __restrict__ A, int lda,
        const unsigned short* __restrict__ Bm, int ldb,
        int rowBase, int colB, int K,
        unsigned short* As, unsigned int* Bp,
        v8f acc[4][2]) {
    const int tid  = threadIdx.x;
    const int lane = tid & 31;
    const int wave = tid >> 5;
    const int wm = wave >> 2, wn = wave & 3;
    const int hi = lane >> 4, nl = lane & 15;
    const int ksteps = K >> 5;

    // staging coordinates (one A-unit pair + one B pair-unit per thread)
    const int arow0 = tid >> 2,          a16_0 = (tid & 3) * 8;
    const int arow1 = (tid + 256) >> 2,  a16_1 = ((tid + 256) & 3) * 8;
    const int kp    = tid >> 4;          // 0..15 k-pair
    const int cg    = (tid & 15) * 8;    // column group start

    const unsigned short* aptr0 = A + (size_t)(rowBase + arow0) * lda + a16_0;
    const unsigned short* aptr1 = A + (size_t)(rowBase + arow1) * lda + a16_1;
    const unsigned short* bptr0 = Bm + (size_t)(2 * kp) * ldb + colB + cg;
    const unsigned short* bptr1 = bptr0 + ldb;

    uint4 ra0 = *(const uint4*)(aptr0);
    uint4 ra1 = *(const uint4*)(aptr1);
    uint4 rb0 = *(const uint4*)(bptr0);
    uint4 rb1 = *(const uint4*)(bptr1);

    for (int kt = 0; kt < ksteps; ++kt) {
        // ---- commit staged registers to LDS -------------------------------
        *(uint4*)(As + arow0 * 40 + a16_0) = ra0;
        *(uint4*)(As + arow1 * 40 + a16_1) = ra1;
        uint4 w0, w1;   // interleave rows (2kp, 2kp+1) into k-pair words
        w0.x = (rb0.x & 0xffffu) | (rb1.x << 16);
        w0.y = (rb0.x >> 16)     | (rb1.x & 0xffff0000u);
        w0.z = (rb0.y & 0xffffu) | (rb1.y << 16);
        w0.w = (rb0.y >> 16)     | (rb1.y & 0xffff0000u);
        w1.x = (rb0.z & 0xffffu) | (rb1.z << 16);
        w1.y = (rb0.z >> 16)     | (rb1.z & 0xffff0000u);
        w1.z = (rb0.w & 0xffffu) | (rb1.w << 16);
        w1.w = (rb0.w >> 16)     | (rb1.w & 0xffff0000u);
        *(uint4*)(Bp + kp * 132 + cg)     = w0;
        *(uint4*)(Bp + kp * 132 + cg + 4) = w1;
        __syncthreads();

        // ---- issue next tile's global loads (overlap with WMMA) -----------
        if (kt + 1 < ksteps) {
            ra0 = *(const uint4*)(aptr0 + (kt + 1) * 32);
            ra1 = *(const uint4*)(aptr1 + (kt + 1) * 32);
            rb0 = *(const uint4*)(bptr0 + (size_t)(kt + 1) * 32 * ldb);
            rb1 = *(const uint4*)(bptr1 + (size_t)(kt + 1) * 32 * ldb);
            if (kt + 2 < ksteps) {   // global_prefetch_b8 for K-tile +2
                __builtin_prefetch(aptr0 + (kt + 2) * 32, 0, 1);
                __builtin_prefetch(bptr0 + (size_t)(kt + 2) * 32 * ldb, 0, 1);
            }
        }

        // ---- fragments ----------------------------------------------------
        FragU aU[4];
#pragma unroll
        for (int mt = 0; mt < 4; ++mt) {
            int m = wm * 64 + mt * 16 + nl;
#pragma unroll
            for (int v = 0; v < 8; ++v) {
                int kk = ((v < 4) ? 2 * v : 16 + 2 * (v - 4)) + hi * 8;
                aU[mt].w[v] = *(const unsigned int*)(As + m * 40 + kk);
            }
        }
        FragU bU[2];
#pragma unroll
        for (int nt = 0; nt < 2; ++nt) {
            int n = wn * 32 + nt * 16 + nl;
#pragma unroll
            for (int v = 0; v < 8; ++v)
                bU[nt].w[v] = Bp[((v < 4) ? hi * 4 + v : 8 + hi * 4 + (v - 4)) * 132 + n];
        }

        // ---- 8 WMMAs per K-step -------------------------------------------
#pragma unroll
        for (int mt = 0; mt < 4; ++mt)
#pragma unroll
            for (int nt = 0; nt < 2; ++nt)
                acc[mt][nt] = __builtin_amdgcn_wmma_f32_16x16x32_bf16(
                    false, aU[mt].v, false, bU[nt].v,
                    (short)0, acc[mt][nt], false, false);
        __syncthreads();
    }
}

// ---------------------------------------------------------------------------
// GEMM 1: G = x @ Wt (fast or slow half by column tile).
// Epilogue: val = sigmoid(G)*h ; rowss += val^2 (shfl-reduced) ;
//           merged = bf16(val * norm_w)
// ---------------------------------------------------------------------------
__global__ __launch_bounds__(256) void k_gemm_merge(
        const unsigned short* __restrict__ xbf,
        const unsigned short* __restrict__ fastWt,
        const unsigned short* __restrict__ slowWt,
        const float* __restrict__ hf, const float* __restrict__ hs,
        const float* __restrict__ norm_w,
        unsigned short* __restrict__ merged,
        float* __restrict__ rowss) {
    __shared__ unsigned short As[128 * 40];
    __shared__ unsigned int   Bp[16 * 132];
    __shared__ float rowsq[128];
    __shared__ float nsh[128];

    int rowBase = blockIdx.y * 128;
    int colBase = blockIdx.x * 128;
    bool slow = colBase >= DIM;
    const unsigned short* Bm = slow ? slowWt : fastWt;
    const float* hsel        = slow ? hs : hf;
    int ncoff = slow ? colBase - DIM : colBase;

    if (threadIdx.x < 128) {
        rowsq[threadIdx.x] = 0.0f;
        nsh[threadIdx.x]   = norm_w[colBase + threadIdx.x];
    }

    v8f zero = {0.f, 0.f, 0.f, 0.f, 0.f, 0.f, 0.f, 0.f};
    v8f acc[4][2];
#pragma unroll
    for (int mt = 0; mt < 4; ++mt)
#pragma unroll
        for (int nt = 0; nt < 2; ++nt) acc[mt][nt] = zero;

    gemm_core(xbf, DIM, Bm, DIM, rowBase, ncoff, DIM, As, Bp, acc);

    const int lane = threadIdx.x & 31, wave = threadIdx.x >> 5;
    const int wm = wave >> 2, wn = wave & 3;
    const int hi = lane >> 4, nl = lane & 15;
#pragma unroll
    for (int mt = 0; mt < 4; ++mt)
#pragma unroll
        for (int r = 0; r < 8; ++r) {
            int lrow = wm * 64 + mt * 16 + hi * 8 + r;
            int grow = rowBase + lrow;
            float s = 0.0f;
#pragma unroll
            for (int nt = 0; nt < 2; ++nt) {
                int lcol = wn * 32 + nt * 16 + nl;
                float val = sigf(acc[mt][nt][r]) *
                            hsel[(size_t)grow * DIM + ncoff + lcol];
                s += val * val;
                merged[(size_t)grow * M2 + colBase + lcol] = f2bf(val * nsh[lcol]);
            }
            // rows are uniform across the 16 lanes of each half -> tree reduce
            s += __shfl_xor(s, 1, 32);
            s += __shfl_xor(s, 2, 32);
            s += __shfl_xor(s, 4, 32);
            s += __shfl_xor(s, 8, 32);
            if (nl == 0) atomicAdd(&rowsq[lrow], s);
        }
    __syncthreads();
    if (threadIdx.x < 128)
        atomicAdd(&rowss[rowBase + threadIdx.x], rowsq[threadIdx.x]);
}

// ---------------------------------------------------------------------------
// GEMM 2: out = rs[row] * (merged_bf16 @ mergeWt)   (norm_w already folded)
// ---------------------------------------------------------------------------
__global__ __launch_bounds__(256) void k_gemm_out(
        const unsigned short* __restrict__ merged,
        const unsigned short* __restrict__ mergeWt,
        const float* __restrict__ rs,
        float* __restrict__ out) {
    __shared__ unsigned short As[128 * 40];
    __shared__ unsigned int   Bp[16 * 132];

    int rowBase = blockIdx.y * 128;
    int colBase = blockIdx.x * 128;

    v8f zero = {0.f, 0.f, 0.f, 0.f, 0.f, 0.f, 0.f, 0.f};
    v8f acc[4][2];
#pragma unroll
    for (int mt = 0; mt < 4; ++mt)
#pragma unroll
        for (int nt = 0; nt < 2; ++nt) acc[mt][nt] = zero;

    gemm_core(merged, M2, mergeWt, DIM, rowBase, colBase, M2, As, Bp, acc);

    const int lane = threadIdx.x & 31, wave = threadIdx.x >> 5;
    const int wm = wave >> 2, wn = wave & 3;
    const int hi = lane >> 4, nl = lane & 15;
#pragma unroll
    for (int mt = 0; mt < 4; ++mt)
#pragma unroll
        for (int nt = 0; nt < 2; ++nt)
#pragma unroll
            for (int r = 0; r < 8; ++r) {
                int grow = rowBase + wm * 64 + mt * 16 + hi * 8 + r;
                int gcol = colBase + wn * 32 + nt * 16 + nl;
                out[(size_t)grow * DIM + gcol] = rs[grow] * acc[mt][nt][r];
            }
}

// ---------------------------------------------------------------------------
extern "C" void kernel_launch(void* const* d_in, const int* in_sizes, int n_in,
                              void* d_out, int out_size, void* d_ws, size_t ws_size,
                              hipStream_t stream) {
    (void)in_sizes; (void)n_in; (void)out_size; (void)ws_size;

    const float* x   = (const float*)d_in[0];
    const float* fgs = (const float*)d_in[1];
    const float* fvs = (const float*)d_in[2];
    const float* fow = (const float*)d_in[3];
    const float* sgs = (const float*)d_in[4];
    const float* svs = (const float*)d_in[5];
    const float* sow = (const float*)d_in[6];
    const float* mw  = (const float*)d_in[7];
    const float* nw  = (const float*)d_in[8];
    float* out = (float*)d_out;

    // workspace layout (256B aligned slabs)
    char* w = (char*)d_ws;
    size_t o = 0;
    auto alloc = [&](size_t bytes) {
        size_t r = o;
        o += (bytes + 255) & ~(size_t)255;
        return r;
    };
    const size_t BTD4 = (size_t)BT * DIM * 4;
    size_t o_xbf  = alloc((size_t)BT * DIM * 2);
    size_t o_ff   = alloc(BTD4);
    size_t o_vf   = alloc(BTD4);
    size_t o_fs   = alloc(BTD4);
    size_t o_vs   = alloc(BTD4);
    size_t o_hf   = alloc(BTD4);
    size_t o_hs   = alloc(BTD4);
    size_t o_fWt  = alloc((size_t)DIM * DIM * 2);
    size_t o_sWt  = alloc((size_t)DIM * DIM * 2);
    size_t o_mWt  = alloc((size_t)M2 * DIM * 2);
    size_t o_rss  = alloc((size_t)BT * 4);
    size_t o_chA  = alloc((size_t)2 * NCHK * BD * 4);
    size_t o_chH  = alloc((size_t)2 * NCHK * BD * 4);
    size_t o_car  = alloc((size_t)2 * NCHK * BD * 4);
    size_t o_mrg  = o_ff;  // alias: ff/vf dead before GEMM1 writes merged

    unsigned short* xbf  = (unsigned short*)(w + o_xbf);
    float* ff   = (float*)(w + o_ff);
    float* vf   = (float*)(w + o_vf);
    float* fs   = (float*)(w + o_fs);
    float* vs   = (float*)(w + o_vs);
    float* hf   = (float*)(w + o_hf);
    float* hs   = (float*)(w + o_hs);
    unsigned short* fWt = (unsigned short*)(w + o_fWt);
    unsigned short* sWt = (unsigned short*)(w + o_sWt);
    unsigned short* mWt = (unsigned short*)(w + o_mWt);
    float* rowss = (float*)(w + o_rss);
    float* chA   = (float*)(w + o_chA);
    float* chH   = (float*)(w + o_chH);
    float* carry = (float*)(w + o_car);
    unsigned short* merged = (unsigned short*)(w + o_mrg);

    // 1) weight transposes -> bf16 [K][N]
    k_transpose_bf16<<<dim3(32, 32), 256, 0, stream>>>(fow, fWt, DIM, DIM);
    k_transpose_bf16<<<dim3(32, 32), 256, 0, stream>>>(sow, sWt, DIM, DIM);
    k_transpose_bf16<<<dim3(64, 32), 256, 0, stream>>>(mw, mWt, DIM, M2);

    // 2) pointwise hproj + gates
    k_pointwise<<<(BT * DIM / 2) / 256, 256, 0, stream>>>(
        x, fgs, fvs, sgs, svs, xbf, ff, vf, fs, vs);

    // 3) chunked scans (float2 channel pairs)
    k_chunk<<<(2 * 2048 * NCHK) / 256, 256, 0, stream>>>(ff, vf, fs, vs, chA, chH);
    k_prefix<<<(2 * 2048) / 256, 256, 0, stream>>>(chA, chH, carry);
    k_scanout<<<(2 * 2048 * NCHK) / 256, 256, 0, stream>>>(ff, vf, fs, vs, carry, hf, hs);

    // 4) GEMM1 + gated merge + row sums of squares
    k_zero<<<BT / 256, 256, 0, stream>>>(rowss, BT);
    k_gemm_merge<<<dim3(M2 / 128, BT / 128), 256, 0, stream>>>(
        xbf, fWt, sWt, hf, hs, nw, merged, rowss);

    // 5) row rsqrt scales, then GEMM2 -> fp32 output
    k_rs<<<BT / 256, 256, 0, stream>>>(rowss);
    k_gemm_out<<<dim3(DIM / 128, BT / 128), 256, 0, stream>>>(merged, mWt, rowss, out);
}
